// GCNEncoder1_14705968022273
// MI455X (gfx1250) — matrix operand order repaired and checked by reading, not compiled, not run on hardware
//
#include <hip/hip_runtime.h>

typedef __attribute__((ext_vector_type(16))) __bf16        v16bf;
typedef __attribute__((ext_vector_type(8)))  float         v8f;
typedef __attribute__((ext_vector_type(8)))  unsigned int  v8u;

#define D 256  // D_IN == D_OUT == 256 in the reference

// ---------- helpers ----------
__device__ __forceinline__ unsigned short f32_to_bf16(float f) {
    unsigned int u = __builtin_bit_cast(unsigned int, f);
    unsigned int r = u + 0x7FFFu + ((u >> 16) & 1u);   // round-to-nearest-even
    return (unsigned short)(r >> 16);
}
__device__ __forceinline__ float bf16_to_f32(unsigned int lo16) {
    return __builtin_bit_cast(float, lo16 << 16);
}
__device__ __forceinline__ unsigned int hash_u32(unsigned int z) {
    z += 0x9E3779B9u;
    z ^= z >> 16; z *= 0x85EBCA6Bu;
    z ^= z >> 13; z *= 0xC2B2AE35u;
    z ^= z >> 16;
    return z;
}

// ---------- init ----------
__global__ void k_init_deg(unsigned int* deg, int N) {
    int i = blockIdx.x * blockDim.x + threadIdx.x;
    if (i < N) deg[i] = 1u;                 // self-loop contributes 1 to every node
}
__global__ void k_zero_f32(float* p, long long n) {
    long long i = (long long)blockIdx.x * blockDim.x + threadIdx.x;
    if (i < n) p[i] = 0.0f;
}

// ---------- degree & norm ----------
__global__ void k_degree(const int* __restrict__ ei, unsigned int* deg, int E) {
    int e = blockIdx.x * blockDim.x + threadIdx.x;
    if (e < E) atomicAdd(&deg[ei[E + e]], 1u);   // row 1 of edge_index = dst
}
__global__ void k_dinv(const unsigned int* __restrict__ deg, float* dinv, int N) {
    int i = blockIdx.x * blockDim.x + threadIdx.x;
    if (i < N) dinv[i] = rsqrtf((float)deg[i]);  // deg >= 1 always (self-loop)
}

// ---------- dropout + bf16 convert of x ----------
__global__ void k_dropout_bf16(const float* __restrict__ x,
                               unsigned short* __restrict__ xb, long long n) {
    long long i = (long long)blockIdx.x * blockDim.x + threadIdx.x;
    if (i >= n) return;
    bool keep = hash_u32((unsigned int)i) & 1u;          // p_keep = 0.5
    float v = keep ? x[i] * 2.0f : 0.0f;                 // inverted dropout 1/keep = 2
    xb[i] = f32_to_bf16(v);
}

// ---------- W -> bf16, transposed so B-fragment K pairs are contiguous ----------
__global__ void k_convW(const float* __restrict__ W, unsigned short* __restrict__ wbt) {
    int i = blockIdx.x * blockDim.x + threadIdx.x;       // i over 256*256
    if (i >= D * D) return;
    int k = i / D, n = i % D;                            // W is [K=256][N=256] row-major
    wbt[n * D + k] = f32_to_bf16(W[i]);
}

// ---------- WMMA GEMM: hb[N,256](bf16) = xb[N,256](bf16) @ W(bf16) ----------
// block = 256 threads = 8 waves; grid = (rowTiles, 2); wave w -> n-tile blockIdx.y*8+w
__global__ void k_gemm_wmma(const unsigned short* __restrict__ xb,
                            const unsigned short* __restrict__ wbt,
                            unsigned short* __restrict__ hb, int N) {
    const int wave = threadIdx.x >> 5;
    const int lane = threadIdx.x & 31;
    const int r0   = blockIdx.x * 16;
    const int c0   = (blockIdx.y * 8 + wave) * 16;
    const int m    = lane & 15;
    const int hf   = lane >> 4;

    int rm = r0 + m; if (rm >= N) rm = N - 1;            // tail clamp (stores guarded)
    const unsigned short* arow = xb  + (size_t)rm       * D;
    const unsigned short* bcol = wbt + (size_t)(c0 + m) * D;   // column n of W = row of wbt

    v8f acc = {};
    for (int k0 = 0; k0 < D; k0 += 32) {
        v8u au, bu;
#pragma unroll
        for (int v = 0; v < 8; ++v) {
            // ISA 7.12.2 16-bit A 16x32: VGPR v<4 -> K=2v + 8*half; v>=4 -> +16
            int ka = k0 + ((v & 3) << 1) + (hf << 3) + ((v >> 2) << 4);
            au[v] = *(const unsigned int*)(arow + ka);
            // 16-bit B 32x16: same K mapping as A: VGPR v<4 -> K=2v + 8*half; v>=4 -> +16
            int kb = k0 + ((v & 3) << 1) + (hf << 3) + ((v >> 2) << 4);
            bu[v] = *(const unsigned int*)(bcol + kb);
        }
        v16bf av = __builtin_bit_cast(v16bf, au);
        v16bf bv = __builtin_bit_cast(v16bf, bu);
        acc = __builtin_amdgcn_wmma_f32_16x16x32_bf16(
                  false, av, false, bv, (short)0, acc, false, false);
    }
    // D layout: VGPR r -> M=r (lanes 0-15), M=r+8 (lanes 16-31); N = lane&15
#pragma unroll
    for (int r = 0; r < 8; ++r) {
        int row = r0 + r + (hf << 3);
        if (row < N) hb[(size_t)row * D + c0 + m] = f32_to_bf16(acc[r]);
    }
}

// ---------- wave-per-edge gather->scale->scatter-add (L2-resident atomics) ----------
__global__ void k_aggregate(const int* __restrict__ ei,
                            const unsigned short* __restrict__ hb,
                            const float* __restrict__ dinv,
                            float* __restrict__ out, int E) {
    int e = blockIdx.x * 8 + (threadIdx.x >> 5);
    if (e >= E) return;
    int lane = threadIdx.x & 31;
    int src = ei[e];
    int dst = ei[E + e];
    float nrm = dinv[src] * dinv[dst];

    const uint4 p = *(const uint4*)(hb + (size_t)src * D + lane * 8); // 8 bf16 / lane
    float* orow = out + (size_t)dst * D + lane * 8;
    unsigned int w[4] = {p.x, p.y, p.z, p.w};
#pragma unroll
    for (int j = 0; j < 4; ++j) {
        atomicAdd(orow + 2 * j,     bf16_to_f32(w[j] & 0xFFFFu) * nrm);
        atomicAdd(orow + 2 * j + 1, bf16_to_f32(w[j] >> 16)     * nrm);
    }
}

// ---------- self-loop + bias + ReLU ----------
__global__ void k_finalize(float* __restrict__ out,
                           const unsigned short* __restrict__ hb,
                           const float* __restrict__ dinv,
                           const float* __restrict__ b, long long n) {
    long long i = (long long)blockIdx.x * blockDim.x + threadIdx.x;
    if (i >= n) return;
    int node = (int)(i / D);
    int col  = (int)(i % D);
    float di = dinv[node];
    float v = out[i] + bf16_to_f32((unsigned int)hb[i]) * di * di + b[col];
    out[i] = v > 0.0f ? v : 0.0f;
}

// ---------- launcher ----------
extern "C" void kernel_launch(void* const* d_in, const int* in_sizes, int n_in,
                              void* d_out, int out_size, void* d_ws, size_t ws_size,
                              hipStream_t stream) {
    const float* x  = (const float*)d_in[0];
    const int*   ei = (const int*)d_in[1];
    const float* W  = (const float*)d_in[2];
    const float* b  = (const float*)d_in[3];
    float* out = (float*)d_out;

    const int N = in_sizes[0] / D;
    const int E = in_sizes[1] / 2;
    const long long ND = (long long)N * D;

    // workspace layout (256B aligned slices)
    auto align256 = [](size_t v) { return (v + 255) & ~size_t(255); };
    char* ws = (char*)d_ws;
    size_t o = 0;
    unsigned int*   deg  = (unsigned int*)(ws + o); o = align256(o + (size_t)N * 4);
    float*          dinv = (float*)(ws + o);        o = align256(o + (size_t)N * 4);
    unsigned short* wbt  = (unsigned short*)(ws + o); o = align256(o + (size_t)D * D * 2);
    unsigned short* xb   = (unsigned short*)(ws + o); o = align256(o + (size_t)ND * 2);
    unsigned short* hb   = (unsigned short*)(ws + o); o = align256(o + (size_t)ND * 2);

    const int T = 256;
    // 1) init
    k_init_deg<<<(N + T - 1) / T, T, 0, stream>>>(deg, N);
    k_zero_f32<<<(int)((ND + T - 1) / T), T, 0, stream>>>(out, ND);
    // 2) degree + norm
    k_degree<<<(E + T - 1) / T, T, 0, stream>>>(ei, deg, E);
    k_dinv<<<(N + T - 1) / T, T, 0, stream>>>(deg, dinv, N);
    // 3) dropout + convert
    k_dropout_bf16<<<(int)((ND + T - 1) / T), T, 0, stream>>>(x, xb, ND);
    k_convW<<<(D * D + T - 1) / T, T, 0, stream>>>(W, wbt);
    // 4) WMMA GEMM (row tiles x 2 col-groups; 8 waves = 8 n-tiles per block)
    dim3 ggrid((N + 15) / 16, 2);
    k_gemm_wmma<<<ggrid, 256, 0, stream>>>(xb, wbt, hb, N);
    // 5) edge aggregation (wave per edge)
    k_aggregate<<<(E + 7) / 8, 256, 0, stream>>>(ei, hb, dinv, out, E);
    // 6) self-loop + bias + relu
    k_finalize<<<(int)((ND + T - 1) / T), T, 0, stream>>>(out, hb, dinv, b, ND);
}
